// DynamicAttention5_40827959116538
// MI455X (gfx1250) — matrix operand order repaired and checked by reading, not compiled, not run on hardware
//
#include <hip/hip_runtime.h>
#include <hip/hip_bf16.h>

typedef __bf16 bf16;
typedef __attribute__((ext_vector_type(16))) __bf16 v16bf;
typedef __attribute__((ext_vector_type(8)))  __bf16 v8bf;
typedef __attribute__((ext_vector_type(8)))  float  v8f;

#define Q_DIM   512
#define F_DIM   2048
#define H_DIM   512
#define O_DIM   512
#define BB      32
#define LL      128
#define SS      36
#define CAT_DIM 1536          /* 512*3 */
#define ML      (BB*LL)       /* 4096 rows of query/out */
#define MS      (BB*SS)       /* 1152 rows of src/trg   */

// ---------------------------------------------------------------------------
// f32 -> bf16 elementwise convert
// ---------------------------------------------------------------------------
__global__ void cvt_f32_bf16(const float* __restrict__ in, bf16* __restrict__ out, int n) {
    int i = blockIdx.x * blockDim.x + threadIdx.x;
    if (i < n) out[i] = (bf16)in[i];
}

// ---------------------------------------------------------------------------
// W (K x N, f32, row-major) -> WT (N x K, bf16, row-major) via LDS tile
// K, N multiples of 32.
// ---------------------------------------------------------------------------
__global__ void tcvt_f32_bf16(const float* __restrict__ W, bf16* __restrict__ WT,
                              int K, int N) {
    __shared__ float tile[32][33];
    int k0 = blockIdx.y * 32, n0 = blockIdx.x * 32;
    int tx = threadIdx.x, ty = threadIdx.y;          // block (32, 8)
    for (int r = ty; r < 32; r += 8)
        tile[r][tx] = W[(size_t)(k0 + r) * N + (n0 + tx)];
    __syncthreads();
    for (int r = ty; r < 32; r += 8)
        WT[(size_t)(n0 + r) * K + (k0 + tx)] = (bf16)tile[tx][r];
}

// ---------------------------------------------------------------------------
// bf16 WMMA GEMM, LDS-staged + double-buffered.
// C = A(MxK) * B(KxN) + bias, with B supplied transposed: BT (N x K, bf16).
// Block = 256 threads = 8 waves; block tile 128(M) x 128(N).
// Wave grid 4(M) x 2(N): each wave owns 32 x 64 -> 8 WMMAs / 32-K step.
// LDS: A tile 128x32, B tile 128x32, double buffered (32 KB total).
// M % 128 == 0, N % 128 == 0, K % 32 == 0.
// ---------------------------------------------------------------------------
template <bool RELU_F32_OUT>
__global__ __launch_bounds__(256)
void gemm_bf16_wmma(const bf16* __restrict__ A,
                    const bf16* __restrict__ BT,
                    const float* __restrict__ bias,
                    void* __restrict__ Cout,
                    int M, int N, int K) {
    __shared__ __attribute__((aligned(32))) bf16 lds_a[2][128 * 32];
    __shared__ __attribute__((aligned(32))) bf16 lds_b[2][128 * 32];

    const int tid  = threadIdx.x;
    const int lane = tid & 31;
    const int wave = tid >> 5;
    const int wm   = wave >> 1;          // 0..3 : 32-row group
    const int wn   = wave & 1;           // 0..1 : 64-col group
    const int mblk = blockIdx.y * 128;
    const int nblk = blockIdx.x * 128;
    const int mrow = lane & 15;
    const int half = lane >> 4;          // 0 or 1

    // staging coords: each thread copies one 16-element (32B) chunk per tile
    const int srow  = tid >> 1;          // 0..127
    const int skoff = (tid & 1) * 16;    // 0 or 16

    const bf16* gA = A  + (size_t)(mblk + srow) * K + skoff;
    const bf16* gB = BT + (size_t)(nblk + srow) * K + skoff;

    v8f acc[2][4] = {};

    const int nk = K / 32;

    // prologue: stage chunk 0
    *(v16bf*)&lds_a[0][srow * 32 + skoff] = *(const v16bf*)gA;
    *(v16bf*)&lds_b[0][srow * 32 + skoff] = *(const v16bf*)gB;
    __syncthreads();

    for (int kc = 0; kc < nk; ++kc) {
        const int cur = kc & 1, nxt = cur ^ 1;

        if (kc + 2 < nk) {               // pull k+2 chunk toward the caches
            __builtin_prefetch(gA + (size_t)(kc + 2) * 32, 0, 1);
            __builtin_prefetch(gB + (size_t)(kc + 2) * 32, 0, 1);
        }
        if (kc + 1 < nk) {               // stage k+1 chunk into the other buffer
            *(v16bf*)&lds_a[nxt][srow * 32 + skoff] = *(const v16bf*)(gA + (size_t)(kc + 1) * 32);
            *(v16bf*)&lds_b[nxt][srow * 32 + skoff] = *(const v16bf*)(gB + (size_t)(kc + 1) * 32);
        }

        // A fragments: elem e -> K = half*16 + e (contiguous, matches B)
        v16bf af[2];
        #pragma unroll
        for (int ms = 0; ms < 2; ++ms) {
            const bf16* ar = &lds_a[cur][(wm * 32 + ms * 16 + mrow) * 32];
            v8bf lo = *(const v8bf*)(ar + half * 16);
            v8bf hi = *(const v8bf*)(ar + half * 16 + 8);
            #pragma unroll
            for (int e = 0; e < 8; ++e) { af[ms][e] = lo[e]; af[ms][e + 8] = hi[e]; }
        }

        #pragma unroll
        for (int s = 0; s < 4; ++s) {
            // B fragment: lane holds col n = wn*64 + s*16 + (lane&15),
            // K = half*16 + e  -> 16 contiguous bf16 in the LDS row.
            v16bf bfrag = *(const v16bf*)&lds_b[cur][(wn * 64 + s * 16 + mrow) * 32 + half * 16];
            #pragma unroll
            for (int ms = 0; ms < 2; ++ms) {
                acc[ms][s] = __builtin_amdgcn_wmma_f32_16x16x32_bf16(
                    false, af[ms], false, bfrag, (short)0, acc[ms][s], false, false);
            }
        }
        __syncthreads();
    }

    // Epilogue: elem r -> M = base + half*8 + r, N = base + (lane&15)
    #pragma unroll
    for (int ms = 0; ms < 2; ++ms) {
        #pragma unroll
        for (int s = 0; s < 4; ++s) {
            int n = nblk + wn * 64 + s * 16 + mrow;
            float bv = bias[n];
            #pragma unroll
            for (int r = 0; r < 8; ++r) {
                int m = mblk + wm * 32 + ms * 16 + half * 8 + r;
                float v = acc[ms][s][r] + bv;
                if (RELU_F32_OUT) {
                    ((float*)Cout)[(size_t)m * N + n] = v > 0.f ? v : 0.f;
                } else {
                    ((bf16*)Cout)[(size_t)m * N + n] = (bf16)v;
                }
            }
        }
    }
}

// ---------------------------------------------------------------------------
// Fused attention middle.  One block per (b,l) row.
// Factorized softmax:  weight[sig,tau] = pT[sig]*pS[tau]  with
//   S[j] = q . s_key[j], T[i] = q . t_key[i] (both scaled by 0.5/sqrt(512)).
// s_ctx = sum_sig pT[sig]*s_value[sig]   (einsum label transposition in ref)
// t_ctx = sum_tau pS[tau]*t_value[tau]
// Writes X row = [ query_bf16 | s_ctx | t_ctx ]  (ML x 1536 bf16).
// ---------------------------------------------------------------------------
__global__ void attn_middle(const bf16*  __restrict__ qb,    // (ML, 512) bf16
                            const bf16*  __restrict__ skey,  // (MS, 512)
                            const bf16*  __restrict__ tkey,
                            const bf16*  __restrict__ sval,
                            const bf16*  __restrict__ tval,
                            const float* __restrict__ query, // (ML, 512) f32
                            bf16*        __restrict__ X) {   // (ML, 1536)
    const int row = blockIdx.x;            // 0..ML-1
    const int b   = row / LL;
    const int t   = threadIdx.x;           // 256 threads

    __shared__ float qsh[H_DIM];
    __shared__ float pr[2][SS];

    for (int h = t; h < H_DIM; h += 256)
        qsh[h] = (float)qb[(size_t)row * H_DIM + h];
    __syncthreads();

    if (t < 2 * SS) {
        const int which = (t < SS) ? 0 : 1;          // 0 = S scores, 1 = T scores
        const int j     = which ? (t - SS) : t;
        const bf16* key = (which ? tkey : skey) + (size_t)(b * SS + j) * H_DIM;
        float acc = 0.f;
        for (int k = 0; k < H_DIM; ++k) acc += qsh[k] * (float)key[k];
        pr[which][j] = acc * 0.022097086912079608f;  // 0.5 / sqrt(512)
    }
    __syncthreads();

    if (t == 0 || t == 32) {                         // one lane per softmax
        const int which = t ? 1 : 0;
        float mx = -3.0e38f;
        for (int j = 0; j < SS; ++j) mx = fmaxf(mx, pr[which][j]);
        float sum = 0.f;
        for (int j = 0; j < SS; ++j) { float e = __expf(pr[which][j] - mx); pr[which][j] = e; sum += e; }
        float inv = 1.f / sum;
        for (int j = 0; j < SS; ++j) pr[which][j] *= inv;
    }
    __syncthreads();

    const size_t xo = (size_t)row * CAT_DIM;
    for (int h = t; h < H_DIM; h += 256) {
        float cs = 0.f, ct = 0.f;
        #pragma unroll 4
        for (int j = 0; j < SS; ++j) {
            cs += pr[1][j] * (float)sval[(size_t)(b * SS + j) * H_DIM + h]; // pT . s_value
            ct += pr[0][j] * (float)tval[(size_t)(b * SS + j) * H_DIM + h]; // pS . t_value
        }
        X[xo + h]                 = (bf16)query[(size_t)row * Q_DIM + h];
        X[xo + Q_DIM + h]         = (bf16)cs;
        X[xo + Q_DIM + H_DIM + h] = (bf16)ct;
    }
}

// ---------------------------------------------------------------------------
extern "C" void kernel_launch(void* const* d_in, const int* in_sizes, int n_in,
                              void* d_out, int out_size, void* d_ws, size_t ws_size,
                              hipStream_t stream) {
    const float* query = (const float*)d_in[0];
    const float* src   = (const float*)d_in[1];
    const float* trg   = (const float*)d_in[2];
    const float* Wq  = (const float*)d_in[3];  const float* bq  = (const float*)d_in[4];
    const float* Ws  = (const float*)d_in[5];  const float* bs  = (const float*)d_in[6];
    const float* Wt  = (const float*)d_in[7];  const float* bt  = (const float*)d_in[8];
    const float* Wsv = (const float*)d_in[9];  const float* bsv = (const float*)d_in[10];
    const float* Wtv = (const float*)d_in[11]; const float* btv = (const float*)d_in[12];
    const float* Wo  = (const float*)d_in[13]; const float* bo  = (const float*)d_in[14];

    char*  ws  = (char*)d_ws;
    size_t off = 0;
    auto alloc = [&](size_t bytes) -> char* {
        char* p = ws + off;
        off = (off + bytes + 255) & ~(size_t)255;
        return p;
    };

    bf16* qin   = (bf16*)alloc((size_t)ML * Q_DIM * 2);
    bf16* srcb  = (bf16*)alloc((size_t)MS * F_DIM * 2);
    bf16* trgb  = (bf16*)alloc((size_t)MS * F_DIM * 2);
    bf16* WqT   = (bf16*)alloc((size_t)H_DIM * Q_DIM * 2);
    bf16* WsT   = (bf16*)alloc((size_t)H_DIM * F_DIM * 2);
    bf16* WtT   = (bf16*)alloc((size_t)H_DIM * F_DIM * 2);
    bf16* WsvT  = (bf16*)alloc((size_t)H_DIM * F_DIM * 2);
    bf16* WtvT  = (bf16*)alloc((size_t)H_DIM * F_DIM * 2);
    bf16* WoT   = (bf16*)alloc((size_t)O_DIM * CAT_DIM * 2);
    bf16* qproj = (bf16*)alloc((size_t)ML * H_DIM * 2);
    bf16* skey  = (bf16*)alloc((size_t)MS * H_DIM * 2);
    bf16* tkey  = (bf16*)alloc((size_t)MS * H_DIM * 2);
    bf16* sval  = (bf16*)alloc((size_t)MS * H_DIM * 2);
    bf16* tval  = (bf16*)alloc((size_t)MS * H_DIM * 2);
    bf16* X     = (bf16*)alloc((size_t)ML * CAT_DIM * 2);
    (void)ws_size; (void)in_sizes; (void)n_in; (void)out_size;

    // --- stage 0: converts -------------------------------------------------
    {
        int nq = ML * Q_DIM, nf = MS * F_DIM;
        cvt_f32_bf16<<<(nq + 255) / 256, 256, 0, stream>>>(query, qin, nq);
        cvt_f32_bf16<<<(nf + 255) / 256, 256, 0, stream>>>(src,   srcb, nf);
        cvt_f32_bf16<<<(nf + 255) / 256, 256, 0, stream>>>(trg,   trgb, nf);

        dim3 tb(32, 8);
        tcvt_f32_bf16<<<dim3(H_DIM / 32, Q_DIM / 32),   tb, 0, stream>>>(Wq,  WqT,  Q_DIM,   H_DIM);
        tcvt_f32_bf16<<<dim3(H_DIM / 32, F_DIM / 32),   tb, 0, stream>>>(Ws,  WsT,  F_DIM,   H_DIM);
        tcvt_f32_bf16<<<dim3(H_DIM / 32, F_DIM / 32),   tb, 0, stream>>>(Wt,  WtT,  F_DIM,   H_DIM);
        tcvt_f32_bf16<<<dim3(H_DIM / 32, F_DIM / 32),   tb, 0, stream>>>(Wsv, WsvT, F_DIM,   H_DIM);
        tcvt_f32_bf16<<<dim3(H_DIM / 32, F_DIM / 32),   tb, 0, stream>>>(Wtv, WtvT, F_DIM,   H_DIM);
        tcvt_f32_bf16<<<dim3(O_DIM / 32, CAT_DIM / 32), tb, 0, stream>>>(Wo,  WoT,  CAT_DIM, O_DIM);
    }

    // --- stage 1: projection GEMMs (WMMA bf16, LDS double-buffered) --------
    {
        dim3 blk(256);
        gemm_bf16_wmma<false><<<dim3(H_DIM / 128, ML / 128), blk, 0, stream>>>(
            qin, WqT, bq, qproj, ML, H_DIM, Q_DIM);
        gemm_bf16_wmma<false><<<dim3(H_DIM / 128, MS / 128), blk, 0, stream>>>(
            srcb, WsT, bs, skey, MS, H_DIM, F_DIM);
        gemm_bf16_wmma<false><<<dim3(H_DIM / 128, MS / 128), blk, 0, stream>>>(
            trgb, WtT, bt, tkey, MS, H_DIM, F_DIM);
        gemm_bf16_wmma<false><<<dim3(H_DIM / 128, MS / 128), blk, 0, stream>>>(
            srcb, WsvT, bsv, sval, MS, H_DIM, F_DIM);
        gemm_bf16_wmma<false><<<dim3(H_DIM / 128, MS / 128), blk, 0, stream>>>(
            trgb, WtvT, btv, tval, MS, H_DIM, F_DIM);
    }

    // --- stage 2: fused scores + factorized softmax + contexts + concat ----
    attn_middle<<<dim3(ML), dim3(256), 0, stream>>>(
        qproj, skey, tkey, sval, tval, query, X);

    // --- stage 3: output GEMM + bias + ReLU (f32 out) ----------------------
    gemm_bf16_wmma<true><<<dim3(O_DIM / 128, ML / 128), dim3(256), 0, stream>>>(
        X, WoT, bo, d_out, ML, O_DIM, CAT_DIM);
}
